// Encoder_73426760892700
// MI455X (gfx1250) — compile-verified, matrix operand order checked by reading, not run on hardware
//
#include <hip/hip_runtime.h>
#include <hip/hip_bf16.h>
#include <stdint.h>

// ---------------------------------------------------------------------------
// DA-RNN encoder for MI455X (gfx1250, wave32, WMMA, async global->LDS).
// B=128, T=256, N=256, M=256, 2M+T = 768.
//
//   EX[b][t'][n] = sum_k Wx[t'][k] * X[b][k][n]        (once, WMMA GEMM)
//   per step t:
//     Uh[b][t']  = sum_d hs[b][d] * U[t'][d]           (WMMA GEMM, hs=[h,c])
//     e[b][n]    = sum_t' ve[t'] * tanh(Uh + EX)       (elementwise)
//     alpha      = softmax_n(e);  x~ = x_t * alpha
//     gates      = [x~,h] @ [WihT;WhhT] + bias         (WMMA GEMM)
//     LSTM update -> h,c, out[t] = h
//
// All B operands are pre-swizzled into per-lane WMMA fragment order so a
// B fragment is 2 contiguous ds_load_b128 per lane. Tile staging uses
// global_load_async_to_lds_b128 + s_wait_asynccnt (ASYNCcnt path).
// ---------------------------------------------------------------------------

#define Bsz 128
#define Tlen 256
#define Nn 256
#define Mm 256

typedef __attribute__((ext_vector_type(16))) __bf16 v16bf;
typedef __attribute__((ext_vector_type(8)))  float  v8f;

__device__ __forceinline__ float fast_tanh(float x) {
  return 1.0f - 2.0f / (__expf(2.0f * x) + 1.0f);   // v_exp_f32 based
}
__device__ __forceinline__ float fast_sigmoid(float x) {
  return 1.0f / (1.0f + __expf(-x));
}

// ------------------------- async global->LDS copy --------------------------
// Per ISA 10 / 15.18.3: VDST = per-lane LDS byte offset, VADDR = 64-bit
// global address. Tracked by ASYNCcnt.
__device__ __forceinline__ void async_copy16(void* lds, const void* gaddr) {
  uint32_t lds_off = (uint32_t)(uintptr_t)lds;   // low 32 bits = LDS offset
  asm volatile("global_load_async_to_lds_b128 %0, %1, off"
               :: "v"(lds_off), "v"(gaddr)
               : "memory");
}
__device__ __forceinline__ void wait_async() {
  asm volatile("s_wait_asynccnt 0x0" ::: "memory");
}

// ----------------- B-matrix swizzle (WMMA fragment layout) -----------------
// Logical B[k][n] (K x Ncols) -> flat index so that a 32x16 sub-tile is
// exactly one wave-fragment: [(k/32)*(Ncols/16) + n/16][lane][j], 16 bf16
// contiguous per lane.
__device__ __forceinline__ size_t bsw_idx(int k, int n, int Ncols) {
  int kc = k >> 5, kl = k & 31;
  int half = (kl >> 3) & 1;
  int j = (kl < 16) ? (kl - 8 * half) : (kl - 8 - 8 * half);
  int lane = (n & 15) + (half << 4);
  return (((size_t)kc * (Ncols >> 4) + (n >> 4)) * 32 + lane) * 16 + j;
}

// ---------------- WMMA fragment loads (ISA 7.12.2 layouts) -----------------
// A fragment 16x32 bf16: lane half h holds K in [8h,8h+8) and [16+8h,16+8h+8),
// both contiguous 16-byte runs in a row-major LDS tile.
__device__ __forceinline__ v16bf load_frag_A(const __bf16* lds, int r0, int ld) {
  int lane = threadIdx.x & 31;
  int half = lane >> 4;
  const __bf16* p = lds + (size_t)(r0 + (lane & 15)) * ld + 8 * half;
  v16bf f;
  ((uint4*)&f)[0] = *(const uint4*)p;          // j=0..7  -> k = 8h .. 8h+7
  ((uint4*)&f)[1] = *(const uint4*)(p + 16);   // j=8..15 -> k = 16+8h ..
  return f;
}

// B fragment from swizzled LDS: 32 contiguous bytes per lane.
__device__ __forceinline__ v16bf load_frag_B_sw(const __bf16* lds, int c0) {
  int lane = threadIdx.x & 31;
  const __bf16* p = lds + ((size_t)(c0 >> 4) * 32 + lane) * 16;
  v16bf f;
  ((uint4*)&f)[0] = *(const uint4*)p;
  ((uint4*)&f)[1] = *(const uint4*)(p + 8);
  return f;
}

__device__ __forceinline__ v8f bfmma(v16bf a, v16bf b, v8f c) {
  return __builtin_amdgcn_wmma_f32_16x16x32_bf16(
      false, a, false, b, (short)0, c, false, false);
}

// Wave computes 32x32 output (2x2 of 16x16), K-chunk = 32.
__device__ __forceinline__ void mma_step(const __bf16* As, const __bf16* Bs,
                                         int wm, int wn, v8f acc[2][2]) {
  v16bf a0 = load_frag_A(As, wm * 32,      32);
  v16bf a1 = load_frag_A(As, wm * 32 + 16, 32);
  v16bf b0 = load_frag_B_sw(Bs, wn * 32);
  v16bf b1 = load_frag_B_sw(Bs, wn * 32 + 16);
  acc[0][0] = bfmma(a0, b0, acc[0][0]);
  acc[0][1] = bfmma(a0, b1, acc[0][1]);
  acc[1][0] = bfmma(a1, b0, acc[1][0]);
  acc[1][1] = bfmma(a1, b1, acc[1][1]);
}

// C/D layout: VGPR r -> row M = r + 8*(lane>=16), col N = lane&15.
__device__ __forceinline__ void store_tile(float* Cg, int ldc, int wm, int wn,
                                           v8f acc[2][2]) {
  int lane = threadIdx.x & 31;
  int half = lane >> 4, nl = lane & 15;
#pragma unroll
  for (int ti = 0; ti < 2; ++ti)
#pragma unroll
    for (int tj = 0; tj < 2; ++tj) {
      float* cp = Cg + (size_t)(wm * 32 + ti * 16 + 8 * half) * ldc
                     + wn * 32 + tj * 16 + nl;
#pragma unroll
      for (int r = 0; r < 8; ++r) cp[(size_t)r * ldc] = acc[ti][tj][r];
    }
}

// ------------------------------ prep kernels -------------------------------
__global__ void k_cvt_x(const float* __restrict__ X, __bf16* __restrict__ Xsw) {
  size_t total = (size_t)Bsz * Tlen * Nn;
  for (size_t i = (size_t)blockIdx.x * blockDim.x + threadIdx.x; i < total;
       i += (size_t)gridDim.x * blockDim.x) {
    int b = (int)(i >> 16);
    int k = ((int)i >> 8) & 255;   // time index
    int n = (int)i & 255;          // series index
    Xsw[(size_t)b * Tlen * Nn + bsw_idx(k, n, Nn)] = (__bf16)X[i];
  }
}

__global__ void k_prep_we(const float* __restrict__ WU_e,
                          __bf16* __restrict__ Wx, __bf16* __restrict__ Utsw) {
  int idx = blockIdx.x * 256 + threadIdx.x;            // 256*768 elements
  int tp = idx / 768, d = idx % 768;
  float v = WU_e[idx];
  if (d < 512) Utsw[bsw_idx(d, tp, 256)] = (__bf16)v;          // B: 512x256
  else         Wx[(size_t)tp * 256 + (d - 512)] = (__bf16)v;   // A: row-major
}

__global__ void k_prep_wg(const float* __restrict__ W_ih,
                          const float* __restrict__ W_hh,
                          __bf16* __restrict__ Wgsw) {
  int idx = blockIdx.x * 256 + threadIdx.x;            // 1024*256 elements
  int j = idx >> 8, k = idx & 255;
  Wgsw[bsw_idx(k,       j, 1024)] = (__bf16)W_ih[idx];   // B: 512x1024
  Wgsw[bsw_idx(k + 256, j, 1024)] = (__bf16)W_hh[idx];
}

__global__ void k_prep_bias(const float* __restrict__ b_ih,
                            const float* __restrict__ b_hh,
                            float* __restrict__ bias) {
  int j = blockIdx.x * 256 + threadIdx.x;
  if (j < 1024) bias[j] = b_ih[j] + b_hh[j];
}

__global__ void k_init(float* __restrict__ h, float* __restrict__ c,
                       __bf16* __restrict__ hsbf) {
  int idx = blockIdx.x * 256 + threadIdx.x;            // 128*512
  hsbf[idx] = (__bf16)0.0f;
  if (idx < Bsz * Mm) { h[idx] = 0.0f; c[idx] = 0.0f; }
}

// ----------------------- shared tile staging (async) -----------------------
// A: 64x32 row-major (4KB). B: one swizzled K-chunk x 128 cols (8KB,
// contiguous in global).
__device__ __forceinline__ void stage_A_async(__bf16* As, const __bf16* Ag,
                                              int lda, int k0) {
  int tid = threadIdx.x;
  int r = tid >> 2, c = (tid & 3) * 8;
  async_copy16(As + r * 32 + c, Ag + (size_t)r * lda + k0 + c);
}
__device__ __forceinline__ void stage_B_async(__bf16* Bs, const __bf16* Bg) {
  int c = threadIdx.x * 16;
  async_copy16(Bs + c, Bg + c);
  async_copy16(Bs + c + 8, Bg + c + 8);
}

// ------------------------- EX = Wx @ X[b]  (once) --------------------------
// grid (128, 4, 2): batch b, M-tile(64) over t', N-tile(128) over n.
__global__ __launch_bounds__(256) void k_ex_gemm(
    const __bf16* __restrict__ Wx, const __bf16* __restrict__ Xsw,
    float* __restrict__ EX) {
  int b = blockIdx.x, mt = blockIdx.y, nt = blockIdx.z;
  __shared__ __bf16 As[64 * 32];
  __shared__ __bf16 Bs[32 * 128];
  const __bf16* Ag = Wx + (size_t)(mt * 64) * 256;
  const __bf16* Bb = Xsw + (size_t)b * Tlen * Nn;
  float* Cg = EX + (size_t)b * Tlen * Nn + (size_t)(mt * 64) * 256 + nt * 128;

  int wave = threadIdx.x >> 5;
  int wm = wave >> 2, wn = wave & 3;
  v8f acc[2][2] = {};
  for (int k0 = 0; k0 < 256; k0 += 32) {
    __syncthreads();
    stage_A_async(As, Ag, 256, k0);
    stage_B_async(Bs, Bb + ((size_t)(k0 >> 5) * 16 + nt * 8) * 512);
    wait_async();
    __syncthreads();
    mma_step(As, Bs, wm, wn, acc);
  }
  store_tile(Cg, 256, wm, wn, acc);
}

// --------------------- Uh = hs_bf(128x512) @ Ut(512x256) -------------------
// grid (2, 2): M-tiles(64) over b, N-tiles(128) over t'.
__global__ __launch_bounds__(256) void k_uh_gemm(
    const __bf16* __restrict__ hsbf, const __bf16* __restrict__ Utsw,
    float* __restrict__ Uh) {
  int mt = blockIdx.x, nt = blockIdx.y;
  __shared__ __bf16 As[64 * 32];
  __shared__ __bf16 Bs[32 * 128];
  const __bf16* Ag = hsbf + (size_t)(mt * 64) * 512;
  float* Cg = Uh + (size_t)(mt * 64) * 256 + nt * 128;

  int wave = threadIdx.x >> 5;
  int wm = wave >> 2, wn = wave & 3;
  v8f acc[2][2] = {};
  for (int k0 = 0; k0 < 512; k0 += 32) {
    __syncthreads();
    stage_A_async(As, Ag, 512, k0);
    stage_B_async(Bs, Utsw + ((size_t)(k0 >> 5) * 16 + nt * 8) * 512);
    wait_async();
    __syncthreads();
    mma_step(As, Bs, wm, wn, acc);
  }
  store_tile(Cg, 256, wm, wn, acc);
}

// --------- e partials: epart[chunk][b][n] = sum_{t' in chunk} ve*tanh ------
__global__ __launch_bounds__(256) void k_attn_part(
    const float* __restrict__ EX, const float* __restrict__ Uh,
    const float* __restrict__ ve, float* __restrict__ epart) {
  int b = blockIdx.x, chunk = blockIdx.y, n = threadIdx.x;
  __shared__ float uh[64], vv[64];
  if (n < 64) {
    uh[n] = Uh[(size_t)b * 256 + chunk * 64 + n];
    vv[n] = ve[chunk * 64 + n];
  }
  __syncthreads();
  const float* exb = EX + (size_t)b * Tlen * Nn + (size_t)(chunk * 64) * 256;
  float acc = 0.0f;
#pragma unroll 8
  for (int tp = 0; tp < 64; ++tp)
    acc += vv[tp] * fast_tanh(uh[tp] + exb[(size_t)tp * 256 + n]);
  epart[((size_t)chunk * Bsz + b) * 256 + n] = acc;
}

// ------------- softmax over n, x~ = x_t * alpha  (bf16 for GEMM) -----------
__global__ __launch_bounds__(256) void k_softmax_xt(
    const float* __restrict__ epart, const float* __restrict__ X,
    __bf16* __restrict__ xtbf, int t) {
  int b = blockIdx.x, n = threadIdx.x;
  float e = 0.0f;
#pragma unroll
  for (int p = 0; p < 4; ++p) e += epart[((size_t)p * Bsz + b) * 256 + n];
  __shared__ float red[256];
  red[n] = e; __syncthreads();
  for (int s = 128; s > 0; s >>= 1) {
    if (n < s) red[n] = fmaxf(red[n], red[n + s]);
    __syncthreads();
  }
  float mx = red[0]; __syncthreads();
  float ex = __expf(e - mx);
  red[n] = ex; __syncthreads();
  for (int s = 128; s > 0; s >>= 1) {
    if (n < s) red[n] += red[n + s];
    __syncthreads();
  }
  float alpha = ex / red[0];
  float xv = X[(size_t)b * Tlen * Nn + (size_t)t * Nn + n];
  xtbf[(size_t)b * 256 + n] = (__bf16)(xv * alpha);
}

// ------------- gates = [x~,h](128x512) @ Wg(512x1024) + bias ---------------
// grid (2, 8): M-tiles(64) over b, N-tiles(128) over j.
__global__ __launch_bounds__(256) void k_gates_gemm(
    const __bf16* __restrict__ xtbf, const __bf16* __restrict__ hsbf,
    const __bf16* __restrict__ Wgsw, const float* __restrict__ bias,
    float* __restrict__ gates) {
  int mt = blockIdx.x, nt = blockIdx.y;
  __shared__ __bf16 As[64 * 32];
  __shared__ __bf16 Bs[32 * 128];
  float* Cg = gates + (size_t)(mt * 64) * 1024 + nt * 128;

  int tid = threadIdx.x, wave = tid >> 5, lane = tid & 31;
  int wm = wave >> 2, wn = wave & 3;
  v8f acc[2][2] = {};
  for (int k0 = 0; k0 < 512; k0 += 32) {
    __syncthreads();
    { int r = tid >> 2, c = (tid & 3) * 8;
      int brow = mt * 64 + r;
      const __bf16* src = (k0 < 256)
          ? (xtbf + (size_t)brow * 256 + k0 + c)
          : (hsbf + (size_t)brow * 512 + (k0 - 256) + c);
      async_copy16(As + r * 32 + c, src); }
    stage_B_async(Bs, Wgsw + ((size_t)(k0 >> 5) * 64 + nt * 8) * 512);
    wait_async();
    __syncthreads();
    mma_step(As, Bs, wm, wn, acc);
  }
  int half = lane >> 4, nl = lane & 15;
#pragma unroll
  for (int ti = 0; ti < 2; ++ti)
#pragma unroll
    for (int tj = 0; tj < 2; ++tj) {
      float bv = bias[nt * 128 + wn * 32 + tj * 16 + nl];
      float* cp = Cg + (size_t)(wm * 32 + ti * 16 + 8 * half) * 1024
                     + wn * 32 + tj * 16 + nl;
#pragma unroll
      for (int r = 0; r < 8; ++r) cp[(size_t)r * 1024] = acc[ti][tj][r] + bv;
    }
}

// ----------------------------- LSTM update ---------------------------------
__global__ __launch_bounds__(256) void k_lstm(
    const float* __restrict__ gates, float* __restrict__ h,
    float* __restrict__ c, __bf16* __restrict__ hsbf,
    float* __restrict__ out_t) {
  int b = blockIdx.x, m = threadIdx.x;
  const float* g = gates + (size_t)b * 1024;
  float gi = fast_sigmoid(g[m]);
  float gf = fast_sigmoid(g[256 + m]);
  float gg = fast_tanh(g[512 + m]);
  float go = fast_sigmoid(g[768 + m]);
  size_t idx = (size_t)b * 256 + m;
  float cn = gf * c[idx] + gi * gg;
  float hn = go * fast_tanh(cn);
  c[idx] = cn;
  h[idx] = hn;
  hsbf[(size_t)b * 512 + m]       = (__bf16)hn;
  hsbf[(size_t)b * 512 + 256 + m] = (__bf16)cn;
  out_t[idx] = hn;
}

// ------------------------------- launcher ----------------------------------
extern "C" void kernel_launch(void* const* d_in, const int* in_sizes, int n_in,
                              void* d_out, int out_size, void* d_ws,
                              size_t ws_size, hipStream_t stream) {
  const float* X    = (const float*)d_in[0];
  const float* WU_e = (const float*)d_in[1];
  const float* v_e  = (const float*)d_in[2];
  const float* W_ih = (const float*)d_in[3];
  const float* W_hh = (const float*)d_in[4];
  const float* b_ih = (const float*)d_in[5];
  const float* b_hh = (const float*)d_in[6];
  float* out = (float*)d_out;

  char* ws = (char*)d_ws;
  size_t o = 0;
  auto take = [&](size_t bytes) -> char* {
    char* p = ws + o;
    o += (bytes + 255) & ~(size_t)255;
    return p;
  };
  float*  EX    = (float*)  take((size_t)Bsz * Tlen * Nn * 4);   // 33.5 MB
  __bf16* Xsw   = (__bf16*) take((size_t)Bsz * Tlen * Nn * 2);   // 16.8 MB
  __bf16* Wx    = (__bf16*) take(256 * 256 * 2);
  __bf16* Utsw  = (__bf16*) take(512 * 256 * 2);
  __bf16* Wgsw  = (__bf16*) take(512 * 1024 * 2);
  float*  bias  = (float*)  take(1024 * 4);
  float*  Uh    = (float*)  take(128 * 256 * 4);
  float*  hbuf  = (float*)  take(128 * 256 * 4);
  float*  cbuf  = (float*)  take(128 * 256 * 4);
  __bf16* hsbf  = (__bf16*) take(128 * 512 * 2);
  __bf16* xtbf  = (__bf16*) take(128 * 256 * 2);
  float*  gates = (float*)  take(128 * 1024 * 4);
  float*  epart = (float*)  take(4 * 128 * 256 * 4);
  (void)ws_size; (void)in_sizes; (void)n_in; (void)out_size;

  // one-time prep
  k_cvt_x<<<4096, 256, 0, stream>>>(X, Xsw);
  k_prep_we<<<768, 256, 0, stream>>>(WU_e, Wx, Utsw);
  k_prep_wg<<<1024, 256, 0, stream>>>(W_ih, W_hh, Wgsw);
  k_prep_bias<<<4, 256, 0, stream>>>(b_ih, b_hh, bias);
  k_init<<<256, 256, 0, stream>>>(hbuf, cbuf, hsbf);
  k_ex_gemm<<<dim3(Bsz, 4, 2), 256, 0, stream>>>(Wx, Xsw, EX);

  // sequential time loop (h/c dependency)
  for (int t = 0; t < Tlen; ++t) {
    k_uh_gemm<<<dim3(2, 2), 256, 0, stream>>>(hsbf, Utsw, Uh);
    k_attn_part<<<dim3(Bsz, 4), 256, 0, stream>>>(EX, Uh, v_e, epart);
    k_softmax_xt<<<Bsz, 256, 0, stream>>>(epart, X, xtbf, t);
    k_gates_gemm<<<dim3(2, 8), 256, 0, stream>>>(xtbf, hsbf, Wgsw, bias,
                                                 gates);
    k_lstm<<<Bsz, 256, 0, stream>>>(gates, hbuf, cbuf, hsbf,
                                    out + (size_t)t * Bsz * Mm);
  }
}
